// Model_48816598286986
// MI455X (gfx1250) — matrix operand order repaired and checked by reading, not compiled, not run on hardware
//
#include <hip/hip_runtime.h>
#include <math.h>

#define NUM_USERS  100000
#define NUM_MOVIES 50000
#define E_SIMS     1500000
#define E_RR       2000000
#define E_LBL      500000

typedef __attribute__((ext_vector_type(16))) _Float16 v16h;
typedef __attribute__((ext_vector_type(8)))  float    v8f;

// ---------------------------------------------------------------------------
// WMMA helper (CDNA5 wave32, v_wmma_f32_16x16x32_f16)
// ---------------------------------------------------------------------------
__device__ __forceinline__ v8f wmma_f16(v16h a, v16h b, v8f c) {
  // (neg_a, A, neg_b, B, c_mod, C, reuse_a, reuse_b)
  return __builtin_amdgcn_wmma_f32_16x16x32_f16(false, a, false, b, (short)0, c,
                                                false, false);
}

// A fragment: 16 rows x 32 K of a row-major [*,64] f32 matrix, converted to
// f16. ISA layout: lane l -> row = l%16; hi = l/16; halves 0..7 =
// K[kbase+8*hi .. +7], halves 8..15 = K[kbase+16+8*hi .. +7]. 4x b128 loads.
__device__ __forceinline__ v16h load_a64(const float* __restrict__ X, int row,
                                         int kbase, int hi) {
  const float* p = X + (size_t)row * 64 + kbase + 8 * hi;
  float t[16];
  *(float4*)(t + 0)  = *(const float4*)(p + 0);
  *(float4*)(t + 4)  = *(const float4*)(p + 4);
  *(float4*)(t + 8)  = *(const float4*)(p + 16);
  *(float4*)(t + 12) = *(const float4*)(p + 20);
  v16h a;
#pragma unroll
  for (int j = 0; j < 16; ++j) a[j] = (_Float16)t[j];
  return a;
}

// B fragment from the pre-packed f16 weight image: fragment (nt,kk) is stored
// lane-major, 16 halves (32 B) per lane -> one aligned v16h load (2x b128),
// fully coalesced across the wave.
__device__ __forceinline__ v16h load_bp(const _Float16* __restrict__ Wp,
                                        int ntile, int kk, int lane) {
  const _Float16* p = Wp + ((size_t)((ntile * 2 + kk) * 32 + lane) << 4);
  return *(const v16h*)p;
}

// ---------------------------------------------------------------------------
// Weight pre-pack: f32 [64,64] row-major -> f16 wave32 B-fragment layout.
// packed[(nt*2+kk)*32*16 + lane*16 + j] = (f16)W[32*kk + 8*(lane>>4) + j + 8*(j>>3)]
//                                             [nt*16 + (lane&15)]
// ---------------------------------------------------------------------------
struct PackSrc { const float* w[17]; };

__global__ void pack_weights_kernel(PackSrc ps, _Float16* __restrict__ out) {
  const float* __restrict__ W = ps.w[blockIdx.x];
  _Float16* __restrict__ o = out + ((size_t)blockIdx.x << 12);
  for (int i = threadIdx.x; i < 4096; i += blockDim.x) {
    int j    = i & 15;
    int lane = (i >> 4) & 31;
    int kk   = (i >> 9) & 1;
    int nt   = (i >> 10) & 3;
    int n = lane & 15, hi = lane >> 4;
    o[i] = (_Float16)W[(size_t)(32 * kk + 8 * hi + j + 8 * (j >> 3)) * 64 + nt * 16 + n];
  }
}

// ---------------------------------------------------------------------------
// Segment-mean building blocks (HBM-bound path: coalesced float4 gather +
// global_atomic_add_f32 scatter)
// ---------------------------------------------------------------------------
__global__ void count_edges_kernel(const int* __restrict__ dst, int nE,
                                   float* __restrict__ cnt) {
  int e = blockIdx.x * blockDim.x + threadIdx.x;
  if (e < nE) atomicAdd(&cnt[dst[e]], 1.0f);
}

__global__ void seg_scatter_add_kernel(const float* __restrict__ X,
                                       const int* __restrict__ src,
                                       const int* __restrict__ dst, int nE,
                                       float* __restrict__ agg) {
  long long idx = (long long)blockIdx.x * blockDim.x + threadIdx.x;
  if (idx >= (long long)nE * 16) return;
  int e  = (int)(idx >> 4);
  int c4 = (int)(idx & 15) << 2;
  int s = src[e], d = dst[e];
  float4 v = *(const float4*)(X + (size_t)s * 64 + c4);
  float* o = agg + (size_t)d * 64 + c4;
  atomicAdd(o + 0, v.x);
  atomicAdd(o + 1, v.y);
  atomicAdd(o + 2, v.z);
  atomicAdd(o + 3, v.w);
}

__global__ void div_by_cnt_kernel(float* __restrict__ agg,
                                  const float* __restrict__ cnt, int n64) {
  int i = blockIdx.x * blockDim.x + threadIdx.x;
  if (i < n64) agg[i] = agg[i] / fmaxf(cnt[i >> 6], 1.0f);
}

// ---------------------------------------------------------------------------
// Fused SAGE linear: Y = act( X1[idx1?] @ W1 + bias + X2[idx2?] @ W2 )
// K = 64 per operand; N = 64. One wave -> 16 output rows, 8..16 WMMAs.
// ---------------------------------------------------------------------------
__global__ void sage_gemm64_kernel(const float* __restrict__ X1,
                                   const int* __restrict__ idx1,
                                   const _Float16* __restrict__ Wp1,
                                   const float* __restrict__ X2,
                                   const int* __restrict__ idx2,
                                   const _Float16* __restrict__ Wp2,
                                   const float* __restrict__ bias,
                                   float* __restrict__ Y, int nrows,
                                   int do_relu) {
  int wave = threadIdx.x >> 5;
  int lane = threadIdx.x & 31;
  int tile = blockIdx.x * (blockDim.x >> 5) + wave;
  int base = tile * 16;
  if (base >= nrows) return;  // wave-uniform: EXEC stays all-1 for WMMA

  int m = lane & 15, hi = lane >> 4;
  int r1 = base + m;
  if (idx1) r1 = idx1[r1];
  v16h a1_0 = load_a64(X1, r1, 0, hi);
  v16h a1_1 = load_a64(X1, r1, 32, hi);

  v16h a2_0{}, a2_1{};
  const bool has2 = (X2 != nullptr);
  if (has2) {
    int r2 = base + m;
    if (idx2) r2 = idx2[r2];
    a2_0 = load_a64(X2, r2, 0, hi);
    a2_1 = load_a64(X2, r2, 32, hi);
  }

#pragma unroll
  for (int nt = 0; nt < 4; ++nt) {
    v8f c = {0.f, 0.f, 0.f, 0.f, 0.f, 0.f, 0.f, 0.f};
    c = wmma_f16(a1_0, load_bp(Wp1, nt, 0, lane), c);
    c = wmma_f16(a1_1, load_bp(Wp1, nt, 1, lane), c);
    if (has2) {
      c = wmma_f16(a2_0, load_bp(Wp2, nt, 0, lane), c);
      c = wmma_f16(a2_1, load_bp(Wp2, nt, 1, lane), c);
    }
    float bv = bias[nt * 16 + m];
#pragma unroll
    for (int v = 0; v < 8; ++v) {
      float y = c[v] + bv;  // D layout: row = v + 8*hi, col = nt*16 + (lane&15)
      if (do_relu) y = fmaxf(y, 0.f);
      Y[(size_t)(base + v + 8 * hi) * 64 + nt * 16 + m] = y;
    }
  }
}

// ---------------------------------------------------------------------------
// Fused decoder: out[e] = relu(concat(zu[lu[e]], zm[lm[e]]) @ W1 + b1) @ w2 + b2
// concat along K -> zu@W1[0:64] + zm@W1[64:128] (packed slots lo/hi).
// Final 64->1 projection fused via cross-lane shfl_xor reduction.
// ---------------------------------------------------------------------------
__global__ void decoder64_kernel(const float* __restrict__ zu,
                                 const float* __restrict__ zm,
                                 const int* __restrict__ lu,
                                 const int* __restrict__ lm,
                                 const _Float16* __restrict__ Wp_lo,
                                 const _Float16* __restrict__ Wp_hi,
                                 const float* __restrict__ b1,  // [64]
                                 const float* __restrict__ w2,  // [64]
                                 const float* __restrict__ b2,  // [1]
                                 float* __restrict__ out, int nE) {
  int wave = threadIdx.x >> 5;
  int lane = threadIdx.x & 31;
  int tile = blockIdx.x * (blockDim.x >> 5) + wave;
  int base = tile * 16;
  if (base >= nE) return;

  int m = lane & 15, hi = lane >> 4;
  int ru = lu[base + m];
  int rm = lm[base + m];
  v16h au0 = load_a64(zu, ru, 0, hi);
  v16h au1 = load_a64(zu, ru, 32, hi);
  v16h am0 = load_a64(zm, rm, 0, hi);
  v16h am1 = load_a64(zm, rm, 32, hi);

  float acc[8] = {0.f, 0.f, 0.f, 0.f, 0.f, 0.f, 0.f, 0.f};
#pragma unroll
  for (int nt = 0; nt < 4; ++nt) {
    v8f c = {0.f, 0.f, 0.f, 0.f, 0.f, 0.f, 0.f, 0.f};
    c = wmma_f16(au0, load_bp(Wp_lo, nt, 0, lane), c);
    c = wmma_f16(au1, load_bp(Wp_lo, nt, 1, lane), c);
    c = wmma_f16(am0, load_bp(Wp_hi, nt, 0, lane), c);
    c = wmma_f16(am1, load_bp(Wp_hi, nt, 1, lane), c);
    float bv = b1[nt * 16 + m];
    float wv = w2[nt * 16 + m];
#pragma unroll
    for (int v = 0; v < 8; ++v) {
      float h = fmaxf(c[v] + bv, 0.f);  // hidden[row=v+8*hi][unit=nt*16+m]
      acc[v] += h * wv;
    }
  }
  // Reduce the 16 hidden-unit columns within each half-wave (xor of the low
  // 4 lane bits never leaves the 16-lane group).
#pragma unroll
  for (int off = 1; off < 16; off <<= 1) {
#pragma unroll
    for (int v = 0; v < 8; ++v) acc[v] += __shfl_xor(acc[v], off, 32);
  }
  if (m == 0) {
    float bb = b2[0];
#pragma unroll
    for (int v = 0; v < 8; ++v) out[base + v + 8 * hi] = acc[v] + bb;
  }
}

// ---------------------------------------------------------------------------
// Host orchestration
// ---------------------------------------------------------------------------
static inline int cdiv_ll(long long a, long long b) { return (int)((a + b - 1) / b); }

static void gemm64(const float* X1, const int* i1, const _Float16* Wp1,
                   const float* X2, const int* i2, const _Float16* Wp2,
                   const float* b, float* Y, int nrows, int relu,
                   hipStream_t s) {
  int tiles = nrows / 16;
  int blocks = (tiles + 7) / 8;  // 256 threads = 8 waves per block
  sage_gemm64_kernel<<<blocks, 256, 0, s>>>(X1, i1, Wp1, X2, i2, Wp2, b, Y,
                                            nrows, relu);
}

static void segmean(const float* X, const int* src, const int* dst, int nE,
                    const float* cnt, float* agg, int nrows, hipStream_t s) {
  hipMemsetAsync(agg, 0, (size_t)nrows * 64 * sizeof(float), s);
  seg_scatter_add_kernel<<<cdiv_ll((long long)nE * 16, 256), 256, 0, s>>>(
      X, src, dst, nE, agg);
  div_by_cnt_kernel<<<cdiv_ll((long long)nrows * 64, 256), 256, 0, s>>>(
      agg, cnt, nrows * 64);
}

extern "C" void kernel_launch(void* const* d_in, const int* in_sizes, int n_in,
                              void* d_out, int out_size, void* d_ws,
                              size_t ws_size, hipStream_t stream) {
  // ---- inputs (setup_inputs order) ----
  const float* x_movie  = (const float*)d_in[0];
  const int*   x_user   = (const int*)d_in[1];
  const int*   s_src    = (const int*)d_in[2];
  const int*   s_dst    = s_src + E_SIMS;  // edge_sims is [2, E_SIMS]
  const int*   rr_src   = (const int*)d_in[3];
  const int*   rr_dst   = (const int*)d_in[4];
  const int*   lbl_user = (const int*)d_in[5];
  const int*   lbl_mov  = (const int*)d_in[6];
  const float* user_emb = (const float*)d_in[7];
  int p = 8;
  const float* it_Wl1 = (const float*)d_in[p++]; const float* it_bl1 = (const float*)d_in[p++];
  const float* it_Wr1 = (const float*)d_in[p++]; const float* it_W1  = (const float*)d_in[p++];
  const float* it_b1  = (const float*)d_in[p++]; const float* it_Wl2 = (const float*)d_in[p++];
  const float* it_bl2 = (const float*)d_in[p++]; const float* it_Wr2 = (const float*)d_in[p++];
  const float* it_W2  = (const float*)d_in[p++]; const float* it_b2  = (const float*)d_in[p++];
  const float* u_Wl1  = (const float*)d_in[p++]; const float* u_bl1  = (const float*)d_in[p++];
  const float* u_Wr1  = (const float*)d_in[p++]; const float* u_W1   = (const float*)d_in[p++];
  const float* u_b1   = (const float*)d_in[p++]; const float* u_Wl2  = (const float*)d_in[p++];
  const float* u_bl2  = (const float*)d_in[p++]; const float* u_Wr2  = (const float*)d_in[p++];
  const float* u_W2   = (const float*)d_in[p++]; const float* u_b2   = (const float*)d_in[p++];
  const float* u_Wl3  = (const float*)d_in[p++]; const float* u_bl3  = (const float*)d_in[p++];
  const float* u_Wr3  = (const float*)d_in[p++]; const float* u_W3   = (const float*)d_in[p++];
  const float* u_b3   = (const float*)d_in[p++]; const float* d_W1   = (const float*)d_in[p++];
  const float* d_b1   = (const float*)d_in[p++]; const float* d_W2   = (const float*)d_in[p++];
  const float* d_b2   = (const float*)d_in[p++];

  // ---- workspace layout ----
  const size_t NM64 = (size_t)NUM_MOVIES * 64;
  const size_t NU64 = (size_t)NUM_USERS * 64;
  float* w = (float*)d_ws;
  float* cnt_m = w; w += NUM_MOVIES;
  float* cnt_u = w; w += NUM_USERS;
  float* m0 = w; w += NM64;
  float* m1 = w; w += NM64;
  float* m2 = w; w += NM64;
  float* m3 = w; w += NM64;
  float* u0 = w; w += NU64;
  float* u1 = w; w += NU64;
  float* u2 = w; w += NU64;
  _Float16* wpack = (_Float16*)w;  // 17 * 4096 f16 = 136 KB, 32B-aligned
  float* out = (float*)d_out;

  // ---- pre-pack all 64x64 weight operands into WMMA B-fragment layout ----
  PackSrc ps;
  const float* srcs[17] = {it_Wl1, it_Wr1, it_W1, it_Wl2, it_Wr2, it_W2,
                           u_Wl1,  u_Wr1,  u_W1,  u_Wl2,  u_Wr2,  u_W2,
                           u_Wl3,  u_Wr3,  u_W3,
                           d_W1 /*rows 0..63*/, d_W1 + 64 * 64 /*rows 64..127*/};
  for (int i = 0; i < 17; ++i) ps.w[i] = srcs[i];
  pack_weights_kernel<<<17, 256, 0, stream>>>(ps, wpack);
  const _Float16* P = wpack;
  const _Float16 *p_itWl1 = P + 0 * 4096, *p_itWr1 = P + 1 * 4096,
                 *p_itW1  = P + 2 * 4096, *p_itWl2 = P + 3 * 4096,
                 *p_itWr2 = P + 4 * 4096, *p_itW2  = P + 5 * 4096,
                 *p_uWl1  = P + 6 * 4096, *p_uWr1  = P + 7 * 4096,
                 *p_uW1   = P + 8 * 4096, *p_uWl2  = P + 9 * 4096,
                 *p_uWr2  = P + 10 * 4096, *p_uW2  = P + 11 * 4096,
                 *p_uWl3  = P + 12 * 4096, *p_uWr3 = P + 13 * 4096,
                 *p_uW3   = P + 14 * 4096, *p_dW1lo = P + 15 * 4096,
                 *p_dW1hi = P + 16 * 4096;

  // ---- in-degree counts (shared across convs on the same graph) ----
  hipMemsetAsync(cnt_m, 0, NUM_MOVIES * sizeof(float), stream);
  hipMemsetAsync(cnt_u, 0, NUM_USERS * sizeof(float), stream);
  count_edges_kernel<<<cdiv_ll(E_SIMS, 256), 256, 0, stream>>>(s_dst, E_SIMS, cnt_m);
  count_edges_kernel<<<cdiv_ll(E_RR, 256), 256, 0, stream>>>(rr_dst, E_RR, cnt_u);

  // m0 = mean over sims of x_movie  (shared by item-conv1 and user-conv1)
  segmean(x_movie, s_src, s_dst, E_SIMS, cnt_m, m0, NUM_MOVIES, stream);
  // m1 = relu(m0@it_Wl1 + it_bl1 + x_movie@it_Wr1)
  gemm64(m0, nullptr, p_itWl1, x_movie, nullptr, p_itWr1, it_bl1, m1, NUM_MOVIES, 1, stream);
  // m2 = relu(m0@u_Wl1 + u_bl1 + x_movie@u_Wr1)
  gemm64(m0, nullptr, p_uWl1, x_movie, nullptr, p_uWr1, u_bl1, m2, NUM_MOVIES, 1, stream);
  // m0 = relu(m1@it_W1 + it_b1)        (item h after lin1)
  gemm64(m1, nullptr, p_itW1, nullptr, nullptr, nullptr, it_b1, m0, NUM_MOVIES, 1, stream);
  // m1 = relu(m2@u_W1 + u_b1)          (hm)
  gemm64(m2, nullptr, p_uW1, nullptr, nullptr, nullptr, u_b1, m1, NUM_MOVIES, 1, stream);
  // m2 = mean over sims of m0
  segmean(m0, s_src, s_dst, E_SIMS, cnt_m, m2, NUM_MOVIES, stream);
  // m3 = relu(m2@it_Wl2 + it_bl2 + m0@it_Wr2)
  gemm64(m2, nullptr, p_itWl2, m0, nullptr, p_itWr2, it_bl2, m3, NUM_MOVIES, 1, stream);
  // m0 = m3@it_W2 + it_b2              (z_movie)
  gemm64(m3, nullptr, p_itW2, nullptr, nullptr, nullptr, it_b2, m0, NUM_MOVIES, 0, stream);

  // u0 = mean over rr of x_movie
  segmean(x_movie, rr_src, rr_dst, E_RR, cnt_u, u0, NUM_USERS, stream);
  // u1 = relu(u0@u_Wl2 + u_bl2 + user_emb[x_user]@u_Wr2)
  gemm64(u0, nullptr, p_uWl2, user_emb, x_user, p_uWr2, u_bl2, u1, NUM_USERS, 1, stream);
  // u0 = relu(u1@u_W2 + u_b2)          (hu)
  gemm64(u1, nullptr, p_uW2, nullptr, nullptr, nullptr, u_b2, u0, NUM_USERS, 1, stream);
  // u1 = mean over rr of hm (m1)
  segmean(m1, rr_src, rr_dst, E_RR, cnt_u, u1, NUM_USERS, stream);
  // u2 = relu(u1@u_Wl3 + u_bl3 + u0@u_Wr3)
  gemm64(u1, nullptr, p_uWl3, u0, nullptr, p_uWr3, u_bl3, u2, NUM_USERS, 1, stream);
  // u0 = u2@u_W3 + u_b3                (z_user)
  gemm64(u2, nullptr, p_uW3, nullptr, nullptr, nullptr, u_b3, u0, NUM_USERS, 0, stream);

  // fused decoder -> d_out
  {
    int tiles = E_LBL / 16;
    int blocks = (tiles + 7) / 8;
    decoder64_kernel<<<blocks, 256, 0, stream>>>(u0, m0, lbl_user, lbl_mov,
                                                 p_dW1lo, p_dW1hi, d_b1, d_W2,
                                                 d_b2, out, E_LBL);
  }
}
